// MaskedNeighbourGNN_14972255994002
// MI455X (gfx1250) — matrix operand order repaired and checked by reading, not compiled, not run on hardware
//
#include <hip/hip_runtime.h>
#include <hip/hip_bf16.h>

typedef __attribute__((ext_vector_type(16))) _Float16 v16h;
typedef __attribute__((ext_vector_type(8)))  float    v8f;

#define NEG_INF (-__builtin_inff())
#define F32_MIN_REF (-3.4028234663852886e38f)

// Branch-free tanh: tanh(x) = 1 - 2/(1 + exp(2x)).
// __expf -> single v_exp_f32 (no divergent range reduction);
// __builtin_amdgcn_rcpf -> v_rcp_f32. Saturates correctly at +-1, NaN propagates.
__device__ __forceinline__ float fast_tanh(float x) {
  float e = __expf(2.f * x);
  return 1.f - 2.f * __builtin_amdgcn_rcpf(1.f + e);
}

// ---------------------------------------------------------------------------
// WMMA helpers (CDNA5 gfx1250, wave32)
// ---------------------------------------------------------------------------
__device__ __forceinline__ v8f wmma32(v16h a, v16h b, v8f c) {
  // D(16x16 f32) = A(16x32 f16) * B(32x16 f16) + C
  return __builtin_amdgcn_wmma_f32_16x16x32_f16(false, a, false, b, (short)0, c,
                                                false, false);
}

// A-fragment (16x32 f16) from row-major f32, per ISA 16-bit A-matrix layout:
// halves 0..7 : K = (lane>>4)*8 + m ; halves 8..15 : K = 16 + (lane>>4)*8 + (m-8)
__device__ __forceinline__ v16h load_a_f32(const float* __restrict__ A, int lda,
                                           int kb, int lane) {
  const int r = lane & 15, g8 = (lane >> 4) << 3;
  const float* p0 = A + (size_t)r * lda + kb + g8;
  v16h a;
#pragma unroll
  for (int i = 0; i < 8; ++i) {
    a[i]     = (_Float16)p0[i];
    a[8 + i] = (_Float16)p0[16 + i];
  }
  return a;
}

// A-fragment from f16 LDS tile (row-major, stride 128)
__device__ __forceinline__ v16h load_a_lds(const _Float16* __restrict__ S,
                                           int kb, int lane) {
  const int r = lane & 15, g8 = (lane >> 4) << 3;
  const _Float16* p0 = S + r * 128 + kb + g8;
  v16h a;
#pragma unroll
  for (int i = 0; i < 8; ++i) {
    a[i]     = p0[i];
    a[8 + i] = p0[16 + i];
  }
  return a;
}

// B-fragment from pre-packed buffer: 16 contiguous halves (32B) per lane
__device__ __forceinline__ v16h load_b_pk(const _Float16* __restrict__ P,
                                          int tile, int lane) {
  return *(const v16h*)(P + ((size_t)tile * 32 + lane) * 16);
}

// ---------------------------------------------------------------------------
// Weight pre-pack: W (K x N, row-major f32) -> B-fragment order f16.
// tile t = nt*numKt + kt ; element e = lane*16 + m
// k = kt*32 + (m>>3)*16 + (lane>>4)*8 + (m&7) ; n = nt*16 + (lane&15) ; OOB -> 0
// ---------------------------------------------------------------------------
__global__ void pack_b_kernel(const float* __restrict__ W,
                              _Float16* __restrict__ dst, int K, int N,
                              int numKt, int total) {
  int idx = blockIdx.x * 256 + threadIdx.x;
  if (idx >= total) return;
  int t = idx >> 9, e = idx & 511;
  int nt = t / numKt, kt = t - nt * numKt;
  int l = e >> 4, m = e & 15;
  int k = kt * 32 + ((m >> 3) << 4) + ((l >> 4) << 3) + (m & 7);
  int n = nt * 16 + (l & 15);
  float v = (k < K && n < N) ? W[(size_t)k * N + n] : 0.f;
  dst[idx] = (_Float16)v;
}

// uv[g*256 + 0..127]   = msg_w[g] @ a_ego   (so ego_proj@a_ego  == h . v)
// uv[g*256 + 128..255] = msg_w[g] @ a_nbr   (so nbr_proj@a_nbr  == h . u)
__global__ void prep_uv_kernel(const float* __restrict__ msg_w,
                               const float* __restrict__ attn_a,
                               float* __restrict__ uv) {
  int idx = blockIdx.x * 256 + threadIdx.x;
  if (idx >= 512) return;
  int i = idx & 127, which = (idx >> 7) & 1, g = idx >> 8;
  const float* M = msg_w + (size_t)g * 16384 + (size_t)i * 128;
  const float* a = attn_a + g * 256 + which * 128;
  float s = 0.f;
  for (int n = 0; n < 128; ++n) s = fmaf(M[n], a[n], s);
  uv[idx] = s;
}

// ---------------------------------------------------------------------------
// Kernel 1: lane encoder. One wave per block, one item per block.
// item 0..4095 = ego, 4096..36863 = neighbor (b*8+k).
// enc[item][128] f32 written to workspace.
// ---------------------------------------------------------------------------
__global__ __launch_bounds__(32) void encode_kernel(
    const float* __restrict__ lane_features, const float* __restrict__ lane_mask,
    const float* __restrict__ nbr_features, const float* __restrict__ nbr_lane_mask,
    const _Float16* __restrict__ pk, const float* __restrict__ b0,
    const float* __restrict__ b1, const float* __restrict__ b2,
    const float* __restrict__ att_b, const float* __restrict__ att_v,
    float* __restrict__ enc) {
  __shared__ _Float16 hbuf[16 * 128];  // f16 staging (A-operand source)
  __shared__ float ebuf[16 * 128];     // f32 'e' for pooling
  __shared__ float wrow[16];           // softmax weights

  const int lane  = threadIdx.x;
  const int item  = blockIdx.x;
  const int col16 = lane & 15;
  const int rbase = (lane >> 4) << 3;  // D-layout: row = rbase + j, col = col16

  const float* x = (item < 4096) ? lane_features + (size_t)item * 512
                                 : nbr_features + (size_t)(item - 4096) * 512;
  const float* mrow = (item < 4096) ? lane_mask + (size_t)item * 16
                                    : nbr_lane_mask + (size_t)(item - 4096) * 16;

  // ---- layer 0: h = tanh(x @ W0 + b0), x:16x32 (single K chunk) ----
  {
    v16h ax = load_a_f32(x, 32, 0, lane);
#pragma unroll
    for (int nt = 0; nt < 8; ++nt) {
      v8f c = {};
      c = wmma32(ax, load_b_pk(pk + 0, nt, lane), c);
      int col = nt * 16 + col16;
      float bb = b0[col];
#pragma unroll
      for (int j = 0; j < 8; ++j)
        hbuf[(rbase + j) * 128 + col] = (_Float16)fast_tanh(c[j] + bb);
    }
  }
  __syncthreads();

  // ---- layer 1: h = tanh(h @ W1 + b1) ----
  {
    v16h af[4];
#pragma unroll
    for (int kt = 0; kt < 4; ++kt) af[kt] = load_a_lds(hbuf, kt * 32, lane);
#pragma unroll
    for (int nt = 0; nt < 8; ++nt) {
      v8f c = {};
#pragma unroll
      for (int kt = 0; kt < 4; ++kt)
        c = wmma32(af[kt], load_b_pk(pk + 4096, nt * 4 + kt, lane), c);
      int col = nt * 16 + col16;
      float bb = b1[col];
#pragma unroll
      for (int j = 0; j < 8; ++j)
        hbuf[(rbase + j) * 128 + col] = (_Float16)fast_tanh(c[j] + bb);
    }
  }
  __syncthreads();

  // ---- layer 2: e = h @ W2 + b2 (f32 to ebuf, f16 to hbuf) ----
  {
    v16h af[4];
#pragma unroll
    for (int kt = 0; kt < 4; ++kt) af[kt] = load_a_lds(hbuf, kt * 32, lane);
#pragma unroll
    for (int nt = 0; nt < 8; ++nt) {
      v8f c = {};
#pragma unroll
      for (int kt = 0; kt < 4; ++kt)
        c = wmma32(af[kt], load_b_pk(pk + 20480, nt * 4 + kt, lane), c);
      int col = nt * 16 + col16;
      float bb = b2[col];
#pragma unroll
      for (int j = 0; j < 8; ++j) {
        float v = c[j] + bb;
        ebuf[(rbase + j) * 128 + col] = v;
        hbuf[(rbase + j) * 128 + col] = (_Float16)v;
      }
    }
  }
  __syncthreads();

  // ---- attention: s = tanh(e @ att_w + att_b) @ att_v  (16x64 -> 16) ----
  float part[8] = {0.f, 0.f, 0.f, 0.f, 0.f, 0.f, 0.f, 0.f};
  {
    v16h af[4];
#pragma unroll
    for (int kt = 0; kt < 4; ++kt) af[kt] = load_a_lds(hbuf, kt * 32, lane);
#pragma unroll
    for (int nt = 0; nt < 4; ++nt) {
      v8f c = {};
#pragma unroll
      for (int kt = 0; kt < 4; ++kt)
        c = wmma32(af[kt], load_b_pk(pk + 36864, nt * 4 + kt, lane), c);
      int col = nt * 16 + col16;
      float ab = att_b[col], av = att_v[col];
#pragma unroll
      for (int j = 0; j < 8; ++j) part[j] += fast_tanh(c[j] + ab) * av;
    }
  }
  // reduce over 16 lanes inside each half-wave (columns)
#pragma unroll
  for (int off = 1; off < 16; off <<= 1)
#pragma unroll
    for (int j = 0; j < 8; ++j) part[j] += __shfl_xor(part[j], off, 32);

  // masked softmax over the 16 rows
  float sv[8], wv[8];
#pragma unroll
  for (int j = 0; j < 8; ++j)
    sv[j] = (mrow[rbase + j] > 0.f) ? part[j] : NEG_INF;
  float mx = sv[0];
#pragma unroll
  for (int j = 1; j < 8; ++j) mx = fmaxf(mx, sv[j]);
  mx = fmaxf(mx, __shfl_xor(mx, 16, 32));
  float se = 0.f;
#pragma unroll
  for (int j = 0; j < 8; ++j) {
    float e = (sv[j] == NEG_INF) ? 0.f : __expf(sv[j] - mx);
    wv[j] = e;
    se += e;
  }
  se += __shfl_xor(se, 16, 32);
  float inv = (se > 0.f) ? 1.f / se : 0.f;
  if (col16 == 0) {
#pragma unroll
    for (int j = 0; j < 8; ++j) wrow[rbase + j] = wv[j] * inv;
  }
  __syncthreads();

  // ---- weighted pooling over rows -> enc[item][0..127] ----
  {
    int c0 = lane * 4;
    float a0 = 0.f, a1 = 0.f, a2 = 0.f, a3 = 0.f;
#pragma unroll
    for (int r = 0; r < 16; ++r) {
      float wr = wrow[r];
      const float* er = &ebuf[r * 128 + c0];
      a0 = fmaf(wr, er[0], a0);
      a1 = fmaf(wr, er[1], a1);
      a2 = fmaf(wr, er[2], a2);
      a3 = fmaf(wr, er[3], a3);
    }
    float* o = enc + (size_t)item * 128 + c0;
    o[0] = a0; o[1] = a1; o[2] = a2; o[3] = a3;
  }
}

// ---------------------------------------------------------------------------
// Kernel 2: message passing (G=2) + actor/critic heads.
// One wave per block, 16 batch rows per wave.
// ---------------------------------------------------------------------------
__global__ __launch_bounds__(32) void gnn_kernel(
    const float* __restrict__ enc, const float* __restrict__ uv,
    const _Float16* __restrict__ pk, const float* __restrict__ nbr_mask,
    const float* __restrict__ action_mask, const float* __restrict__ self_b,
    const float* __restrict__ actor_b0, const float* __restrict__ actor_b1,
    const float* __restrict__ critic_b0, const float* __restrict__ critic_b1,
    float* __restrict__ out) {
  __shared__ _Float16 hbuf[16 * 128];  // current h_ego tile (f16, row-major)
  __shared__ _Float16 hbar[16 * 128];  // weighted neighbor sums / head hidden

  const int lane  = threadIdx.x;
  const int b0r   = blockIdx.x * 16;
  const int col16 = lane & 15;
  const int rbase = (lane >> 4) << 3;

  // load initial h_ego (16 rows x 128) as f16
  for (int t = 0; t < 64; ++t) {
    int i = t * 32 + lane;
    hbuf[i] = (_Float16)enc[(size_t)b0r * 128 + i];
  }
  __syncthreads();

  const int r  = lane >> 1;        // row 0..15 (pair of lanes per row)
  const int ch = (lane & 1) * 64;  // column half

  for (int g = 0; g < 2; ++g) {
    const float* vg   = uv + g * 256;  // [0..127]=v(ego), [128..255]=u(nbr)
    const float* encN = enc + (size_t)(4096 + (b0r + r) * 8) * 128;

    // ego logit term: h_ego . v
    float le = 0.f;
#pragma unroll
    for (int c2 = 0; c2 < 64; ++c2)
      le = fmaf((float)hbuf[r * 128 + ch + c2], vg[ch + c2], le);
    le += __shfl_xor(le, 1, 32);

    // neighbor logits, leaky relu, mask
    float lg[8];
#pragma unroll
    for (int k = 0; k < 8; ++k) {
      const float* hn = encN + k * 128 + ch;
      float ln = 0.f;
#pragma unroll
      for (int c2 = 0; c2 < 64; ++c2) ln = fmaf(hn[c2], vg[128 + ch + c2], ln);
      ln += __shfl_xor(ln, 1, 32);
      float L = le + ln;
      L = (L >= 0.f) ? L : 0.2f * L;
      lg[k] = (nbr_mask[(b0r + r) * 8 + k] > 0.f) ? L : NEG_INF;
    }
    float mx = lg[0];
#pragma unroll
    for (int k = 1; k < 8; ++k) mx = fmaxf(mx, lg[k]);
    float se = 0.f, wk[8];
#pragma unroll
    for (int k = 0; k < 8; ++k) {
      float e = (lg[k] == NEG_INF) ? 0.f : __expf(lg[k] - mx);
      wk[k] = e;
      se += e;
    }
    float inv = (se > 0.f) ? 1.f / se : 0.f;

    // hbar = sum_k w_k * H_nei[k]  (msg = hbar @ msg_w by linearity)
    float acc[64];
#pragma unroll
    for (int c2 = 0; c2 < 64; ++c2) acc[c2] = 0.f;
#pragma unroll
    for (int k = 0; k < 8; ++k) {
      float w = wk[k] * inv;
      const float* hn = encN + k * 128 + ch;
#pragma unroll
      for (int c2 = 0; c2 < 64; ++c2) acc[c2] = fmaf(w, hn[c2], acc[c2]);
    }
#pragma unroll
    for (int c2 = 0; c2 < 64; ++c2)
      hbar[r * 128 + ch + c2] = (_Float16)acc[c2];
    __syncthreads();

    // h_ego = tanh(hbar@msg_w[g] + h_ego@self_w[g] + self_b[g])
    v16h am[4], as[4];
#pragma unroll
    for (int kt = 0; kt < 4; ++kt) {
      am[kt] = load_a_lds(hbar, kt * 32, lane);
      as[kt] = load_a_lds(hbuf, kt * 32, lane);
    }
    const _Float16* pm = pk + 45056 + g * 16384;
    const _Float16* ps = pk + 77824 + g * 16384;
#pragma unroll
    for (int nt = 0; nt < 8; ++nt) {
      v8f c = {};
#pragma unroll
      for (int kt = 0; kt < 4; ++kt)
        c = wmma32(am[kt], load_b_pk(pm, nt * 4 + kt, lane), c);
#pragma unroll
      for (int kt = 0; kt < 4; ++kt)
        c = wmma32(as[kt], load_b_pk(ps, nt * 4 + kt, lane), c);
      int col = nt * 16 + col16;
      float bb = self_b[g * 128 + col];
#pragma unroll
      for (int j = 0; j < 8; ++j)
        hbuf[(rbase + j) * 128 + col] = (_Float16)fast_tanh(c[j] + bb);
    }
    __syncthreads();
  }

  // ---- heads ----
  v16h ah[4];
#pragma unroll
  for (int kt = 0; kt < 4; ++kt) ah[kt] = load_a_lds(hbuf, kt * 32, lane);

  // actor hidden: h16 = h_ego @ actor_w0 + actor_b0 -> hbar
#pragma unroll
  for (int nt = 0; nt < 8; ++nt) {
    v8f c = {};
#pragma unroll
    for (int kt = 0; kt < 4; ++kt)
      c = wmma32(ah[kt], load_b_pk(pk + 110592, nt * 4 + kt, lane), c);
    int col = nt * 16 + col16;
    float bb = actor_b0[col];
#pragma unroll
    for (int j = 0; j < 8; ++j)
      hbar[(rbase + j) * 128 + col] = (_Float16)(c[j] + bb);
  }
  __syncthreads();
  {
    v16h aa[4];
#pragma unroll
    for (int kt = 0; kt < 4; ++kt) aa[kt] = load_a_lds(hbar, kt * 32, lane);
    v8f c = {};
#pragma unroll
    for (int kt = 0; kt < 4; ++kt)
      c = wmma32(aa[kt], load_b_pk(pk + 126976, kt, lane), c);  // N padded to 16
    if (col16 < 8) {
#pragma unroll
      for (int j = 0; j < 8; ++j) {
        int b = b0r + rbase + j;
        float v = c[j] + actor_b1[col16];
        const float* am_ = action_mask + (size_t)b * 8;
        bool alli = true;
#pragma unroll
        for (int a = 0; a < 8; ++a) alli = alli && (am_[a] <= 0.f);
        float m = (am_[col16] <= 0.f) ? F32_MIN_REF : v;
        out[(size_t)b * 8 + col16] = alli ? v : m;
      }
    }
  }
  __syncthreads();

  // critic hidden -> hbar
#pragma unroll
  for (int nt = 0; nt < 8; ++nt) {
    v8f c = {};
#pragma unroll
    for (int kt = 0; kt < 4; ++kt)
      c = wmma32(ah[kt], load_b_pk(pk + 129024, nt * 4 + kt, lane), c);
    int col = nt * 16 + col16;
    float bb = critic_b0[col];
#pragma unroll
    for (int j = 0; j < 8; ++j)
      hbar[(rbase + j) * 128 + col] = (_Float16)(c[j] + bb);
  }
  __syncthreads();
  {
    v16h aa[4];
#pragma unroll
    for (int kt = 0; kt < 4; ++kt) aa[kt] = load_a_lds(hbar, kt * 32, lane);
    v8f c = {};
#pragma unroll
    for (int kt = 0; kt < 4; ++kt)
      c = wmma32(aa[kt], load_b_pk(pk + 145408, kt, lane), c);  // N padded to 16
    if (col16 == 0) {
#pragma unroll
      for (int j = 0; j < 8; ++j)
        out[32768 + b0r + rbase + j] = c[j] + critic_b1[0];
    }
  }
}

// ---------------------------------------------------------------------------
// Host side
// ---------------------------------------------------------------------------
extern "C" void kernel_launch(void* const* d_in, const int* in_sizes, int n_in,
                              void* d_out, int out_size, void* d_ws,
                              size_t ws_size, hipStream_t stream) {
  (void)in_sizes; (void)n_in; (void)out_size; (void)ws_size;
  const float* lane_features = (const float*)d_in[0];
  const float* lane_mask     = (const float*)d_in[1];
  const float* action_mask   = (const float*)d_in[2];
  const float* nbr_features  = (const float*)d_in[3];
  const float* nbr_lane_mask = (const float*)d_in[4];
  const float* nbr_mask      = (const float*)d_in[5];
  const float* mlp_w0 = (const float*)d_in[6];
  const float* mlp_b0 = (const float*)d_in[7];
  const float* mlp_w1 = (const float*)d_in[8];
  const float* mlp_b1 = (const float*)d_in[9];
  const float* mlp_w2 = (const float*)d_in[10];
  const float* mlp_b2 = (const float*)d_in[11];
  const float* att_w  = (const float*)d_in[12];
  const float* att_b  = (const float*)d_in[13];
  const float* att_v  = (const float*)d_in[14];
  const float* msg_w  = (const float*)d_in[15];
  const float* attn_a = (const float*)d_in[16];
  const float* self_w = (const float*)d_in[17];
  const float* self_b = (const float*)d_in[18];
  const float* actor_w0  = (const float*)d_in[19];
  const float* actor_b0  = (const float*)d_in[20];
  const float* actor_w1  = (const float*)d_in[21];
  const float* actor_b1  = (const float*)d_in[22];
  const float* critic_w0 = (const float*)d_in[23];
  const float* critic_b0 = (const float*)d_in[24];
  const float* critic_w1 = (const float*)d_in[25];
  const float* critic_b1 = (const float*)d_in[26];
  float* out = (float*)d_out;

  // workspace layout
  char* ws = (char*)d_ws;
  float*    enc = (float*)ws;                             // 36864*128 f32
  _Float16* pk  = (_Float16*)(ws + 18874368);             // 147456 halves
  float*    uv  = (float*)(ws + 18874368 + 294912);       // 512 f32

  auto pack = [&](const float* W, int off, int K, int N, int numKt, int numNt) {
    int total = numKt * numNt * 512;
    pack_b_kernel<<<(total + 255) / 256, 256, 0, stream>>>(W, pk + off, K, N,
                                                           numKt, total);
  };
  pack(mlp_w0, 0, 32, 128, 1, 8);
  pack(mlp_w1, 4096, 128, 128, 4, 8);
  pack(mlp_w2, 20480, 128, 128, 4, 8);
  pack(att_w, 36864, 128, 64, 4, 4);
  pack(msg_w, 45056, 128, 128, 4, 8);
  pack(msg_w + 16384, 61440, 128, 128, 4, 8);
  pack(self_w, 77824, 128, 128, 4, 8);
  pack(self_w + 16384, 94208, 128, 128, 4, 8);
  pack(actor_w0, 110592, 128, 128, 4, 8);
  pack(actor_w1, 126976, 128, 8, 4, 1);
  pack(critic_w0, 129024, 128, 128, 4, 8);
  pack(critic_w1, 145408, 128, 1, 4, 1);
  prep_uv_kernel<<<2, 256, 0, stream>>>(msg_w, attn_a, uv);

  encode_kernel<<<36864, 32, 0, stream>>>(lane_features, lane_mask, nbr_features,
                                          nbr_lane_mask, pk, mlp_b0, mlp_b1,
                                          mlp_b2, att_b, att_v, enc);

  gnn_kernel<<<256, 32, 0, stream>>>(enc, uv, pk, nbr_mask, action_mask, self_b,
                                     actor_b0, actor_b1, critic_b0, critic_b1,
                                     out);
}
